// CohortExploitationModule_7584912244842
// MI455X (gfx1250) — compile-verified, matrix operand order checked by reading, not run on hardware
//
#include <hip/hip_runtime.h>

// Problem constants (from reference)
#define Bn  256
#define Tn  16
#define Fn  8
#define Dn  32
#define Kn  64
#define Cn  1024
#define On  2
#define BTn 4096       // B*T
#define D1  33         // D+1
#define PD  48         // D1 padded to 3 WMMA n-tiles

typedef __attribute__((ext_vector_type(16))) _Float16 v16h;
typedef __attribute__((ext_vector_type(8)))  _Float16 v8h;
typedef __attribute__((ext_vector_type(8)))  float    v8f;
typedef __attribute__((ext_vector_type(4)))  unsigned int u32x4;
typedef __attribute__((ext_vector_type(8)))  int      i32x8;
typedef __attribute__((ext_vector_type(4)))  int      i32x4;
typedef unsigned long long u64;
typedef unsigned char      u8;
typedef _Float16           h16;

// ---------------- workspace layout (bytes) ----------------
#define WS_CODES  0          // u8 [BT*F]   (32768) read as u64[BT]
#define WS_PATV   32768      // u64[F*C]    (65536)
#define WS_WILD   98304      // u64[F*C]    (65536)
#define WS_KC     163840     // h16[F*C*D]  (524288)
#define WS_VCT    688128     // h16[F*PD*C] (786432) d-major, padded
#define WS_IQ     1474560    // h16[F*B*D]  (131072)
#define WS_BDOT   1605632    // f32[F*B]    (8192)
#define WS_FM     1613824    // f32[B*F]    (8192)
#define WS_MATCH  1622016    // u8 [F*B*C]  (2097152)  total ~3.72 MB

// ---- TDM: 1-D tile (nelem x 2-byte elements) global -> LDS, per ISA D# layout ----
// group0: count=1 | lds_addr | global_addr[56:0] | type=2
// group1: data_size=2B, tensor_dim0=nelem, tensor_dim1=1, tile_dim0=nelem, tile_dim1=1,
//         tensor_dim0_stride=nelem (single row)
__device__ __forceinline__ void tdm_load_to_lds(unsigned lds_off, const void* gptr, unsigned nelem) {
  unsigned long long ga = (unsigned long long)gptr;
  u32x4 g0;
  g0[0] = 1u;                                            // count=1, user descriptor
  g0[1] = lds_off;                                       // lds_addr (bytes)
  g0[2] = (unsigned)ga;                                  // global_addr[31:0]
  g0[3] = (unsigned)((ga >> 32) & 0x01FFFFFFull) | (2u << 30); // addr[56:32] | type=2
  i32x8 g1;
  g1[0] = (int)(1u << 16);                               // wg_mask=0, data_size=1 (2 bytes)
  g1[1] = (int)((nelem & 0xFFFFu) << 16);                // tensor_dim0[15:0] @ bits63:48
  g1[2] = (int)(((nelem >> 16) & 0xFFFFu) | (1u << 16)); // tensor_dim0[31:16] | tensor_dim1=1
  g1[3] = (int)((nelem & 0xFFFFu) << 16);                // tile_dim0 @ bits127:112
  g1[4] = 1;                                             // tile_dim1=1, tile_dim2=0
  g1[5] = (int)nelem;                                    // tensor_dim0_stride[31:0]
  g1[6] = 0;
  g1[7] = 0;
  i32x4 gz = {0, 0, 0, 0};                               // groups 2/3 unused (<=2D)
  i32x8 gz8 = {0, 0, 0, 0, 0, 0, 0, 0};
  __builtin_amdgcn_tensor_load_to_lds(g0, g1, gz, gz, gz8, 0);
}

// -------- stage 1: vector-quantize each (bt, f) row --------
__global__ void k_codes(const float* __restrict__ tdata, const int* __restrict__ f_mask,
                        const float* __restrict__ centers, u8* __restrict__ codes) {
  int idx = blockIdx.x * blockDim.x + threadIdx.x;
  if (idx >= BTn * Fn) return;
  int bt = idx >> 3, f = idx & 7;
  const float* x = tdata + ((size_t)bt * Fn + f) * Dn;
  float xl[Dn];
  #pragma unroll
  for (int j = 0; j < Dn; ++j) xl[j] = x[j];
  int best = 0; float bestd = 3.4e38f;
  for (int k = 0; k < Kn; ++k) {
    const float* c = centers + ((size_t)(f * Kn + k)) * Dn;
    float cc = 0.f, xc = 0.f;
    #pragma unroll
    for (int j = 0; j < Dn; ++j) { float cv = c[j]; cc += cv * cv; xc += xl[j] * cv; }
    float d = cc - 2.0f * xc;
    if (d < bestd) { bestd = d; best = k; }
  }
  codes[idx] = (u8)(f_mask[idx] != 0 ? best + 2 : 1);
}

// -------- stage 2: per-(b,f) "any over T" mask --------
__global__ void k_fm(const int* __restrict__ f_mask, float* __restrict__ fm) {
  int idx = blockIdx.x * blockDim.x + threadIdx.x;
  if (idx >= Bn * Fn) return;
  int b = idx / Fn, f = idx % Fn;
  int any = 0;
  for (int t = 0; t < Tn; ++t) any |= f_mask[((size_t)(b * Tn + t)) * Fn + f];
  fm[idx] = any ? 1.0f : 0.0f;
}

// -------- stage 3: per-pattern precompute --------
__global__ void k_pat(const int* __restrict__ pat, const float* __restrict__ pat_rep,
                      const float* __restrict__ pos, const float* __restrict__ neg,
                      const float* __restrict__ Wk, const float* __restrict__ Wv,
                      u64* __restrict__ patv, u64* __restrict__ wildm,
                      h16* __restrict__ Kc_h, h16* __restrict__ Vct_h) {
  int idx = blockIdx.x * blockDim.x + threadIdx.x;
  if (idx >= Fn * Cn) return;
  int f = idx / Cn, c = idx % Cn;
  u64 pv = 0, wl = 0;
  #pragma unroll
  for (int j = 0; j < Fn; ++j) {
    int p = pat[((size_t)idx) * Fn + j];
    pv |= ((u64)(u8)p) << (8 * j);
    if (p == 0) wl |= ((u64)0xFFu) << (8 * j);
  }
  patv[idx] = pv; wildm[idx] = wl;
  float rep[D1];
  #pragma unroll
  for (int j = 0; j < Dn; ++j) rep[j] = pat_rep[((size_t)idx) * Dn + j];
  float pc = pos[idx], nc = neg[idx];
  rep[Dn] = pc / (pc + nc + 1e-6f);
  for (int d = 0; d < Dn; ++d) {
    const float* w = Wk + ((size_t)(f * Dn + d)) * D1;
    float s = 0.f;
    #pragma unroll
    for (int j = 0; j < D1; ++j) s += rep[j] * w[j];
    Kc_h[((size_t)idx) * Dn + d] = (h16)s;
  }
  for (int dd = 0; dd < PD; ++dd) {
    float s = 0.f;
    if (dd < D1) {
      const float* w = Wv + ((size_t)(f * D1 + dd)) * D1;
      #pragma unroll
      for (int j = 0; j < D1; ++j) s += rep[j] * w[j];
    }
    Vct_h[((size_t)(f * PD + dd)) * Cn + c] = (h16)s;
  }
}

// -------- stage 4: iq = q@Wq^T + bq (f16), bdot = bk . iq (f32) --------
__global__ void k_iq(const float* __restrict__ tdata, const float* __restrict__ Wq,
                     const float* __restrict__ bq, const float* __restrict__ bk,
                     h16* __restrict__ iq_h, float* __restrict__ bdot) {
  int idx = blockIdx.x * blockDim.x + threadIdx.x;
  if (idx >= Fn * Bn) return;
  int f = idx / Bn, b = idx % Bn;
  const float* q = tdata + (((size_t)(b * Tn + (Tn - 1)) * Fn) + f) * Dn;
  float ql[Dn];
  #pragma unroll
  for (int j = 0; j < Dn; ++j) ql[j] = q[j];
  float bd = 0.f;
  for (int d = 0; d < Dn; ++d) {
    const float* w = Wq + ((size_t)(f * Dn + d)) * Dn;
    float s = bq[f * Dn + d];
    #pragma unroll
    for (int j = 0; j < Dn; ++j) s += ql[j] * w[j];
    iq_h[((size_t)idx) * Dn + d] = (h16)s;
    bd += s * bk[f * Dn + d];
  }
  bdot[idx] = bd;
}

// -------- stage 5: branch-free wildcard pattern match, OR over T --------
__global__ void k_match(const u64* __restrict__ cw, const u64* __restrict__ patv,
                        const u64* __restrict__ wildm, u8* __restrict__ match) {
  size_t idx = (size_t)blockIdx.x * blockDim.x + threadIdx.x;
  if (idx >= (size_t)Fn * Bn * Cn) return;
  int c  = (int)(idx % Cn);
  int fb = (int)(idx / Cn);
  int b  = fb % Bn;
  int f  = fb / Bn;
  u64 pv = patv[f * Cn + c];
  u64 nw = ~wildm[f * Cn + c];
  const u64* cr = cw + (size_t)b * Tn;
  int m = 0;
  #pragma unroll
  for (int t = 0; t < Tn; ++t) m |= (((cr[t] ^ pv) & nw) == 0ull);
  match[idx] = (u8)m;
}

// -------- stage 6: WMMA attention with TDM-staged K/V panels --------
__global__ __launch_bounds__(32) void k_attn(
    const h16* __restrict__ iq_h, const float* __restrict__ bdot,
    const h16* __restrict__ Kc_h, const h16* __restrict__ Vct_h,
    const u8* __restrict__ match, const float* __restrict__ bv,
    const float* __restrict__ fm, float* __restrict__ a_all, float* __restrict__ srep) {
  __shared__ float s_e[16 * Cn];   // 64 KB: e tile, then p = a*match
  __shared__ h16  s_kv[PD * Cn];   // 96 KB: Kc panel (64 KB), then Vct panel (96 KB)
  const int lane = threadIdx.x;
  const int f  = blockIdx.x >> 4;
  const int b0 = (blockIdx.x & 15) << 4;
  const int lm = lane & 15;
  const int khi = (lane >> 4) << 3;            // 16-bit A/B striping: K group base
  const int rowbase = b0 + ((lane >> 4) << 3); // D-frag: reg r -> row rowbase+r
  const unsigned lds_kv = (unsigned)(unsigned long long)(const void*)s_kv;

  // TDM: stage this feature's Kc panel (1024x32 f16 = 64 KB) into LDS
  tdm_load_to_lds(lds_kv, Kc_h + (size_t)f * Cn * Dn, Cn * Dn);

  // A fragment: 16x32 f16 iq tile (two 16B loads per lane) — overlaps with TDM
  v16h af;
  {
    const h16* base = iq_h + ((size_t)(f * Bn + b0 + lm)) * Dn + khi;
    v8h lo = *(const v8h*)(base);
    v8h hi = *(const v8h*)(base + 16);
    #pragma unroll
    for (int i = 0; i < 8; ++i) { af[i] = lo[i]; af[i + 8] = hi[i]; }
  }
  __builtin_amdgcn_s_wait_tensorcnt(0);

  // e = iq x Kc^T : 64 n-tiles, one v_wmma_f32_16x16x32_f16 each (K=32 in one shot)
  for (int nt = 0; nt < 64; ++nt) {
    const int c0 = nt << 4;
    const h16* kb = s_kv + (c0 + lm) * Dn + khi;
    v8h lo = *(const v8h*)kb;
    v8h hi = *(const v8h*)(kb + 16);
    v16h bf;
    #pragma unroll
    for (int i = 0; i < 8; ++i) { bf[i] = lo[i]; bf[i + 8] = hi[i]; }
    v8f cz = {};
    v8f e8 = __builtin_amdgcn_wmma_f32_16x16x32_f16(false, af, false, bf, (short)0, cz, false, false);
    #pragma unroll
    for (int r = 0; r < 8; ++r) {
      int row = rowbase + r;
      float bdv = bdot[f * Bn + row];
      u8 mt = match[((size_t)(f * Bn + row)) * Cn + c0 + lm];
      float ev = mt ? (e8[r] + bdv) : (bdv - 1.0e7f);
      s_e[(row - b0) * Cn + c0 + lm] = ev;
    }
  }
  // All Kc LDS reads must have landed before TDM overwrites s_kv with Vct
  asm volatile("s_wait_dscnt 0x0" ::: "memory");
  // TDM: stage Vct panel (48x1024 f16 = 96 KB); overlaps with softmax below
  tdm_load_to_lds(lds_kv, Vct_h + (size_t)f * PD * Cn, PD * Cn);
  __syncthreads();

  // masked softmax over C: lane pair (r, r+16) splits the 1024 columns
  {
    const int r = lm;
    const int seg = (lane >> 4) * 512;
    const int grow = b0 + r;
    float* erow = s_e + r * Cn;
    float mx = -3.0e38f;
    for (int j = 0; j < 512; ++j) mx = fmaxf(mx, erow[seg + j]);
    mx = fmaxf(mx, __shfl_xor(mx, 16, 32));
    float sum = 0.f;
    for (int j = 0; j < 512; ++j) {
      float pe = __expf(erow[seg + j] - mx);
      erow[seg + j] = pe;
      sum += pe;
    }
    sum += __shfl_xor(sum, 16, 32);
    float inv = 1.0f / sum;
    const u8*  mrow = match + ((size_t)(f * Bn + grow)) * Cn;
    float*     arow = a_all + ((size_t)(f * Bn + grow)) * Cn;
    for (int j = 0; j < 512; ++j) {
      float a = erow[seg + j] * inv;
      arow[seg + j] = a;                        // full softmax (a_all output)
      erow[seg + j] = mrow[seg + j] ? a : 0.0f; // p = a * match for v-GEMM
    }
  }
  __builtin_amdgcn_s_wait_tensorcnt(0);         // Vct panel resident
  __syncthreads();

  // v = p x Vct : (16x1024)@(1024x48pad), 32 k-steps x 3 n-tiles of WMMA, B from LDS
  v8f acc[3] = {};
  for (int ks = 0; ks < 32; ++ks) {
    const float* pr = s_e + lm * Cn + (ks << 5) + khi;
    v16h pa;
    #pragma unroll
    for (int i = 0; i < 8; ++i) { pa[i] = (h16)pr[i]; pa[i + 8] = (h16)pr[16 + i]; }
    #pragma unroll
    for (int t = 0; t < 3; ++t) {
      const h16* vb = s_kv + (t * 16 + lm) * Cn + (ks << 5) + khi;
      v8h lo = *(const v8h*)vb;
      v8h hi = *(const v8h*)(vb + 16);
      v16h bf;
      #pragma unroll
      for (int i = 0; i < 8; ++i) { bf[i] = lo[i]; bf[i + 8] = hi[i]; }
      acc[t] = __builtin_amdgcn_wmma_f32_16x16x32_f16(false, pa, false, bf, (short)0, acc[t], false, false);
    }
  }

  // epilogue: v + bv, masked by fm -> srep (B,F,33)
  #pragma unroll
  for (int t = 0; t < 3; ++t) {
    int dd = t * 16 + lm;
    if (dd < D1) {
      float bvv = bv[f * D1 + dd];
      #pragma unroll
      for (int r = 0; r < 8; ++r) {
        int row = rowbase + r;
        float v = acc[t][r] + bvv;
        srep[((size_t)row * Fn + f) * D1 + dd] = v * fm[row * Fn + f];
      }
    }
  }
}

// -------- stage 7: cali = srep.reshape(B,264) @ W_pred^T --------
__global__ void k_pred(const float* __restrict__ srep, const float* __restrict__ Wp,
                       float* __restrict__ cali) {
  int idx = blockIdx.x * blockDim.x + threadIdx.x;
  if (idx >= Bn * On) return;
  int b = idx >> 1, o = idx & 1;
  float s = 0.f;
  for (int i = 0; i < Fn * D1; ++i) s += srep[b * Fn * D1 + i] * Wp[o * Fn * D1 + i];
  cali[b * On + o] = s;
}

extern "C" void kernel_launch(void* const* d_in, const int* in_sizes, int n_in,
                              void* d_out, int out_size, void* d_ws, size_t ws_size,
                              hipStream_t stream) {
  (void)in_sizes; (void)n_in; (void)out_size; (void)ws_size;
  const float* tdata   = (const float*)d_in[0];
  const int*   f_mask  = (const int*)  d_in[1];
  const float* centers = (const float*)d_in[2];
  const int*   pat     = (const int*)  d_in[3];
  const float* pat_rep = (const float*)d_in[4];
  const float* pos_cnt = (const float*)d_in[5];
  const float* neg_cnt = (const float*)d_in[6];
  const float* W_q     = (const float*)d_in[7];
  const float* b_q     = (const float*)d_in[8];
  const float* W_k     = (const float*)d_in[9];
  const float* b_k     = (const float*)d_in[10];
  const float* W_v     = (const float*)d_in[11];
  const float* b_v     = (const float*)d_in[12];
  const float* W_pred  = (const float*)d_in[13];

  char* ws = (char*)d_ws;
  u8*   codes = (u8*)  (ws + WS_CODES);
  u64*  patv  = (u64*) (ws + WS_PATV);
  u64*  wildm = (u64*) (ws + WS_WILD);
  h16*  Kc_h  = (h16*) (ws + WS_KC);
  h16*  Vct_h = (h16*) (ws + WS_VCT);
  h16*  iq_h  = (h16*) (ws + WS_IQ);
  float* bdot = (float*)(ws + WS_BDOT);
  float* fm   = (float*)(ws + WS_FM);
  u8*   match = (u8*)  (ws + WS_MATCH);

  float* out   = (float*)d_out;
  float* cali  = out;                          // (B, O)
  float* a_all = out + (size_t)Bn * On;        // (F, B, C)
  float* srep  = a_all + (size_t)Fn * Bn * Cn; // (B, F, 33)

  k_codes<<<(BTn * Fn + 255) / 256, 256, 0, stream>>>(tdata, f_mask, centers, codes);
  k_fm   <<<(Bn * Fn + 255) / 256, 256, 0, stream>>>(f_mask, fm);
  k_pat  <<<(Fn * Cn + 255) / 256, 256, 0, stream>>>(pat, pat_rep, pos_cnt, neg_cnt,
                                                     W_k, W_v, patv, wildm, Kc_h, Vct_h);
  k_iq   <<<(Fn * Bn + 255) / 256, 256, 0, stream>>>(tdata, W_q, b_q, b_k, iq_h, bdot);
  k_match<<<((size_t)Fn * Bn * Cn + 255) / 256, 256, 0, stream>>>((const u64*)codes, patv, wildm, match);
  k_attn <<<Fn * (Bn / 16), 32, 0, stream>>>(iq_h, bdot, Kc_h, Vct_h, match, b_v, fm, a_all, srep);
  k_pred <<<(Bn * On + 255) / 256, 256, 0, stream>>>(srep, W_pred, cali);
}
